// ContextModule_62706522522373
// MI455X (gfx1250) — matrix operand order repaired and checked by reading, not compiled, not run on hardware
//
#include <hip/hip_runtime.h>
#include <hip/hip_bf16.h>

// ---------------------------------------------------------------------------
// ContextModule for MI455X (gfx1250, wave32, WMMA).
// Heavy GEMMs run as v_wmma_f32_16x16x32_bf16 with fp32 accumulation.
// D=512, B=8, T=1024, H=4, dk=128, N=2000, TOPK=101.
// Stage-1 scores computed for batch 0 only (all the reference uses).
// ---------------------------------------------------------------------------

typedef __attribute__((ext_vector_type(16))) __bf16          v16bf;
typedef __attribute__((ext_vector_type(16))) short           v16s;
typedef __attribute__((ext_vector_type(8)))  float           v8f;
typedef __attribute__((ext_vector_type(8)))  unsigned short  u16x8;
typedef __attribute__((address_space(3))) unsigned short*    lds_us_ptr;

struct V16pair { u16x8 lo, hi; };   // 32B == one v16bf fragment

__device__ __forceinline__ unsigned short f2bf(float f) {
  __bf16 b = (__bf16)f;                       // native RNE convert on gfx1250
  return __builtin_bit_cast(unsigned short, b);
}

__device__ __forceinline__ u16x8 pack8(float4 a, float4 b) {
  u16x8 r;
  r[0] = f2bf(a.x); r[1] = f2bf(a.y); r[2] = f2bf(a.z); r[3] = f2bf(a.w);
  r[4] = f2bf(b.x); r[5] = f2bf(b.y); r[6] = f2bf(b.z); r[7] = f2bf(b.w);
  return r;
}

// A fragment: per-lane 32B contiguous run of 16 k values
__device__ __forceinline__ v16bf ldfragA(const unsigned short* p) {
  V16pair t;
  t.lo = *(const u16x8*)(p);
  t.hi = *(const u16x8*)(p + 8);
  return __builtin_bit_cast(v16bf, t);
}
// B fragment: per-lane 32B contiguous run of 16 k values
__device__ __forceinline__ v16bf ldfragB(const unsigned short* p) {
  V16pair t;
  t.lo = *(const u16x8*)(p);
  t.hi = *(const u16x8*)(p + 8);
  return __builtin_bit_cast(v16bf, t);
}

// CDNA5 async global->LDS copy, 16B per lane, tracked by ASYNCcnt (ISA 10.7)
__device__ __forceinline__ void async_copy_b128(void* lds_dst, const void* gsrc) {
  lds_us_ptr l = (lds_us_ptr)lds_dst;         // addrspace(3): 32-bit LDS address
  asm volatile("global_load_async_to_lds_b128 %0, %1, off"
               :: "v"(l), "v"(gsrc) : "memory");
}
__device__ __forceinline__ void wait_async0() {
  asm volatile("s_wait_asynccnt 0" ::: "memory");
}

// ---------------------------------------------------------------------------
// NT GEMM: C[M,N] = (A[M,K] * W[N,K]^T) * alpha + bias[N]
// fp32 in, bf16 WMMA, fp32 out.  Requires K % 32 == 0 (holds: 512 / 128).
// Block = 128 threads = 4 waves; block tile 64x64; each wave owns a 2x2 set
// of 16x16 C tiles -> 4 WMMA per 32-k chunk with fragment reuse.
// Uniform full-tile fast path: no exec-mask guards on the hot launches.
// ---------------------------------------------------------------------------
__global__ __launch_bounds__(128)
void gemm_nt_wmma(const float* __restrict__ A, int lda,
                  const float* __restrict__ W, int ldw,
                  const float* __restrict__ bias,
                  float* __restrict__ C, int ldc,
                  int M, int N, int K, float alpha)
{
  __shared__ __align__(16) unsigned short As[64][40];  // 80B pitch: rows 16B aligned
  __shared__ __align__(16) unsigned short Ws[64][40];

  const int tid  = threadIdx.x;
  const int lane = tid & 31;
  const int wave = tid >> 5;
  const int m0 = blockIdx.y * 64;
  const int n0 = blockIdx.x * 64;

  const int half = lane >> 4;
  const int l15  = lane & 15;
  const int kbA  = half * 16;     // A: k 0-15 vs 16-31
  const int kbB  = half * 16;     // B: k 0-15 vs 16-31
  const int wm   = (wave >> 1) * 32;
  const int wn   = (wave & 1)  * 32;

  const int lr = tid >> 1;             // 0..63: tile row this thread stages
  const int lh = (tid & 1) << 4;       // 0 | 16: which 16-float half of the chunk

  const bool fullTile = (m0 + 64 <= M) && (n0 + 64 <= N);   // uniform across block
  const float* Arow = A + (size_t)(m0 + lr) * lda + lh;
  const float* Wrow = W + (size_t)(n0 + lr) * ldw + lh;

  v8f acc[2][2] = {};

  for (int k0 = 0; k0 < K; k0 += 32) {
    float4 a0{}, a1{}, a2{}, a3{}, w0{}, w1{}, w2{}, w3{};
    if (fullTile) {                       // scalar branch: no exec masking
      const float4* pa = (const float4*)(Arow + k0);
      const float4* pw = (const float4*)(Wrow + k0);
      a0 = pa[0]; a1 = pa[1]; a2 = pa[2]; a3 = pa[3];
      w0 = pw[0]; w1 = pw[1]; w2 = pw[2]; w3 = pw[3];
    } else {
      if (m0 + lr < M) {
        const float4* pa = (const float4*)(Arow + k0);
        a0 = pa[0]; a1 = pa[1]; a2 = pa[2]; a3 = pa[3];
      }
      if (n0 + lr < N) {
        const float4* pw = (const float4*)(Wrow + k0);
        w0 = pw[0]; w1 = pw[1]; w2 = pw[2]; w3 = pw[3];
      }
    }
    if (k0 + 32 < K) {   // speculative prefetch of next chunk (global_prefetch_b8)
      __builtin_prefetch(Arow + k0 + 32, 0, 0);
      __builtin_prefetch(Wrow + k0 + 32, 0, 0);
    }
    *(u16x8*)&As[lr][lh]     = pack8(a0, a1);
    *(u16x8*)&As[lr][lh + 8] = pack8(a2, a3);
    *(u16x8*)&Ws[lr][lh]     = pack8(w0, w1);
    *(u16x8*)&Ws[lr][lh + 8] = pack8(w2, w3);
    __syncthreads();

    v16bf af[2], bfr[2];
#pragma unroll
    for (int mi = 0; mi < 2; ++mi) af[mi]  = ldfragA(&As[wm + mi * 16 + l15][kbA]);
#pragma unroll
    for (int ni = 0; ni < 2; ++ni) bfr[ni] = ldfragB(&Ws[wn + ni * 16 + l15][kbB]);
#pragma unroll
    for (int mi = 0; mi < 2; ++mi)
#pragma unroll
      for (int ni = 0; ni < 2; ++ni)
        acc[mi][ni] = __builtin_amdgcn_wmma_f32_16x16x32_bf16(
            false, af[mi], false, bfr[ni], (short)0, acc[mi][ni], false, false);
    __syncthreads();
  }

  // C store (ISA 7.12.2 32-bit C layout: vgpr r -> row r / r+8, lane -> col)
  if (fullTile) {
#pragma unroll
    for (int ni = 0; ni < 2; ++ni) {
      const int nG = n0 + wn + ni * 16 + l15;
      const float bz = (bias != nullptr) ? bias[nG] : 0.f;
#pragma unroll
      for (int mi = 0; mi < 2; ++mi)
#pragma unroll
        for (int r = 0; r < 8; ++r) {
          const int mG = m0 + wm + mi * 16 + r + 8 * half;
          C[(size_t)mG * ldc + nG] = acc[mi][ni][r] * alpha + bz;
        }
    }
  } else {
#pragma unroll
    for (int ni = 0; ni < 2; ++ni) {
      const int nG = n0 + wn + ni * 16 + l15;
      const float bz = (bias != nullptr && nG < N) ? bias[nG] : 0.f;
#pragma unroll
      for (int mi = 0; mi < 2; ++mi)
#pragma unroll
        for (int r = 0; r < 8; ++r) {
          const int mG = m0 + wm + mi * 16 + r + 8 * half;
          if (mG < M && nG < N)
            C[(size_t)mG * ldc + nG] = acc[mi][ni][r] * alpha + bz;
        }
    }
  }
}

// ---------------------------------------------------------------------------
// Stage-1 softmax statistics and deterministic column accumulation
// ---------------------------------------------------------------------------
__global__ void init_zero(float* p, int n) {
  int i = blockIdx.x * blockDim.x + threadIdx.x;
  if (i < n) p[i] = 0.f;
}

__global__ __launch_bounds__(256)
void rowstats_kernel(const float* __restrict__ S, int cols,
                     float* __restrict__ rmax, float* __restrict__ rsum)
{
  __shared__ float red[256];
  const int row = blockIdx.x, tid = threadIdx.x;
  const float* p = S + (size_t)row * cols;

  float m = -3.0e38f;
  for (int c = tid; c < cols; c += 256) m = fmaxf(m, p[c]);
  red[tid] = m; __syncthreads();
  for (int s = 128; s > 0; s >>= 1) { if (tid < s) red[tid] = fmaxf(red[tid], red[tid + s]); __syncthreads(); }
  const float rm = red[0]; __syncthreads();

  float su = 0.f;
  for (int c = tid; c < cols; c += 256) su += __expf(p[c] - rm);
  red[tid] = su; __syncthreads();
  for (int s = 128; s > 0; s >>= 1) { if (tid < s) red[tid] += red[tid + s]; __syncthreads(); }
  if (tid == 0) { rmax[row] = rm; rsum[row] = red[0]; }
}

__global__ __launch_bounds__(256)
void colsum_kernel(const float* __restrict__ S, int rows, int cols,
                   const float* __restrict__ rmax, const float* __restrict__ rsum,
                   float* __restrict__ ctxscore)
{
  int c = blockIdx.x * blockDim.x + threadIdx.x;
  if (c >= cols) return;
  float acc = 0.f;
  for (int t = 0; t < rows; ++t)
    acc += __expf(S[(size_t)t * cols + c] - rmax[t]) / rsum[t];
  ctxscore[c] += acc;   // sequential over heads (separate launches) -> deterministic
}

// ---------------------------------------------------------------------------
// Top-101 by iterative argmax (selection order is irrelevant downstream:
// softmax over the selected set is permutation-invariant).
// ---------------------------------------------------------------------------
__global__ __launch_bounds__(256)
void topk_kernel(const float* __restrict__ score, int n, int k, int* __restrict__ out_idx)
{
  __shared__ float vals[2048];
  __shared__ float rv[256];
  __shared__ int   ri[256];
  const int tid = threadIdx.x;
  for (int i = tid; i < 2048; i += 256) vals[i] = (i < n) ? score[i] : -3.0e38f;
  __syncthreads();

  for (int it = 0; it < k; ++it) {
    float bm = -3.0e38f; int bi = 0;
    for (int i = tid; i < n; i += 256)
      if (vals[i] > bm) { bm = vals[i]; bi = i; }
    rv[tid] = bm; ri[tid] = bi; __syncthreads();
    for (int s = 128; s > 0; s >>= 1) {
      if (tid < s) {
        if (rv[tid + s] > rv[tid] ||
            (rv[tid + s] == rv[tid] && ri[tid + s] < ri[tid])) {
          rv[tid] = rv[tid + s]; ri[tid] = ri[tid + s];
        }
      }
      __syncthreads();
    }
    if (tid == 0) { out_idx[it] = ri[0]; vals[ri[0]] = -3.0e38f; }
    __syncthreads();
  }
}

// Gather pre-projected K/V rows for the selected contexts (bf16, padded to 112)
__global__ void gather_kernel(const float* __restrict__ kc, const float* __restrict__ vc,
                              const int* __restrict__ idx,
                              unsigned short* __restrict__ k2, unsigned short* __restrict__ v2)
{
  int e = blockIdx.x * blockDim.x + threadIdx.x;   // 8-element segments: 112*64
  if (e >= 112 * 64) return;
  int r = e >> 6, cs = (e & 63) * 8;
  u16x8 zk = {}, zv = {};
  if (r < 101) {
    int s = idx[r];
    const float4* pk = (const float4*)&kc[(size_t)s * 512 + cs];
    const float4* pv = (const float4*)&vc[(size_t)s * 512 + cs];
    zk = pack8(pk[0], pk[1]);
    zv = pack8(pv[0], pv[1]);
  }
  *(u16x8*)&k2[(size_t)r * 512 + cs] = zk;
  *(u16x8*)&v2[(size_t)r * 512 + cs] = zv;
}

// ---------------------------------------------------------------------------
// Stage-2 fused attention: one wave per (b, h, 16-row t-tile).
// S = Q·K2^T (7 n-tiles x 4 k-chunks WMMA), register softmax with width-16
// shfl_xor reductions matching the C-tile layout, then O = P·V2 (8 d-tiles x
// 4 k-chunks WMMA).  N padded 101 -> 112 (mask) -> 128 (zero P / zero V rows).
// K2/V2 staged into LDS with CDNA5 async copies (ASYNCcnt).
// ---------------------------------------------------------------------------
__global__ __launch_bounds__(32)
void attn2_kernel(const float* __restrict__ q, const unsigned short* __restrict__ k2,
                  const unsigned short* __restrict__ v2, float* __restrict__ o)
{
  __shared__ __align__(16) unsigned short ks[112][128];  // 28 KB
  __shared__ __align__(16) unsigned short vs[128][128];  // 32 KB (rows 112.. zero)
  __shared__ __align__(16) unsigned short ps[16][128];   //  4 KB (total 64 KB)

  const int lane = threadIdx.x;
  const int t0 = blockIdx.x * 16;
  const int h  = blockIdx.y;
  const int b  = blockIdx.z;
  const float scale = 0.0883883476483184f;   // 1/sqrt(128)

  // stage K2/V2 head slices into LDS: async b128 copies, 16B per lane.
  // e = lane + 32*i -> rows {2i, 2i+1}: the r<112 split is uniform per i.
  for (int i = 0; i < 56; ++i) {             // 112*128/8 = 1792 segments
    int e = lane + 32 * i;
    int r = e >> 4, ds_ = (e & 15) * 8;
    async_copy_b128(&ks[r][ds_], &k2[(size_t)r * 512 + h * 128 + ds_]);
    async_copy_b128(&vs[r][ds_], &v2[(size_t)r * 512 + h * 128 + ds_]);
  }
  u16x8 zseg = {};
  for (int i = 56; i < 64; ++i) {            // zero-pad V rows 112..127
    int e = lane + 32 * i;
    *(u16x8*)&vs[e >> 4][(e & 15) * 8] = zseg;
  }
  for (int i = 0; i < 8; ++i) {              // zero P tile (cols 112..127 stay 0)
    int e = lane + 32 * i;
    *(u16x8*)&ps[e >> 4][(e & 15) * 8] = zseg;
  }
  wait_async0();
  __syncthreads();

  const int half = lane >> 4;
  const int l15  = lane & 15;
  const int kbA  = half * 16;
  const int kbB  = half * 16;
  const size_t qrow = ((size_t)(b * 1024 + t0 + l15)) * 512 + h * 128;

  // ---- S = Q * K2^T ----
  v8f accs[7] = {};
  for (int c = 0; c < 4; ++c) {
    const float4* qp0 = (const float4*)&q[qrow + 32 * c + kbA];
    const float4* qp1 = (const float4*)&q[qrow + 32 * c + kbA + 8];
    V16pair qt;
    qt.lo = pack8(qp0[0], qp0[1]);
    qt.hi = pack8(qp1[0], qp1[1]);
    v16bf afrag = __builtin_bit_cast(v16bf, qt);
#pragma unroll
    for (int nt = 0; nt < 7; ++nt) {
      v16bf bfrag = ldfragB(&ks[nt * 16 + l15][32 * c + kbB]);
      accs[nt] = __builtin_amdgcn_wmma_f32_16x16x32_bf16(
          false, afrag, false, bfrag, (short)0, accs[nt], false, false);
    }
  }

  // ---- row softmax over 101 columns (mask cols >= 101) ----
  float pv[7][8];
#pragma unroll
  for (int r = 0; r < 8; ++r) {
    float m = -3.0e38f;
#pragma unroll
    for (int nt = 0; nt < 7; ++nt) {
      int col = nt * 16 + l15;
      if (col < 101) m = fmaxf(m, accs[nt][r] * scale);
    }
    for (int off = 8; off >= 1; off >>= 1) m = fmaxf(m, __shfl_xor(m, off, 16));
    float su = 0.f;
#pragma unroll
    for (int nt = 0; nt < 7; ++nt) {
      int col = nt * 16 + l15;
      float p = (col < 101) ? __expf(accs[nt][r] * scale - m) : 0.f;
      pv[nt][r] = p;
      su += p;
    }
    for (int off = 8; off >= 1; off >>= 1) su += __shfl_xor(su, off, 16);
    float inv = 1.0f / su;
#pragma unroll
    for (int nt = 0; nt < 7; ++nt) pv[nt][r] *= inv;
  }

  // scatter normalized P into LDS per C-tile layout (cols 0..111)
#pragma unroll
  for (int nt = 0; nt < 7; ++nt)
#pragma unroll
    for (int r = 0; r < 8; ++r)
      ps[r + 8 * half][nt * 16 + l15] = f2bf(pv[nt][r]);
  __syncthreads();

  // ---- O = P * V2  (K = 128; P cols >= 112 are zero, V rows >= 112 are zero)
  v8f acco[8] = {};
  for (int c = 0; c < 4; ++c) {
    v16bf afrag = ldfragA(&ps[l15][32 * c + kbA]);
#pragma unroll
    for (int dt = 0; dt < 8; ++dt) {
      v16s bvv;
      const int d = dt * 16 + l15;
#pragma unroll
      for (int j = 0; j < 16; ++j) bvv[j] = (short)vs[32 * c + kbB + j][d];
      v16bf bfrag = __builtin_bit_cast(v16bf, bvv);
      acco[dt] = __builtin_amdgcn_wmma_f32_16x16x32_bf16(
          false, afrag, false, bfrag, (short)0, acco[dt], false, false);
    }
  }

  // store O head slice
#pragma unroll
  for (int dt = 0; dt < 8; ++dt)
#pragma unroll
    for (int r = 0; r < 8; ++r) {
      int row = t0 + r + 8 * half;
      o[((size_t)(b * 1024 + row)) * 512 + h * 128 + dt * 16 + l15] = acco[dt][r];
    }
}

// ---------------------------------------------------------------------------
// out = LayerNorm(enc + proj) * g + b    (one 128-thread block per row, D=512)
// ---------------------------------------------------------------------------
__global__ __launch_bounds__(128)
void ln_kernel(const float* __restrict__ enc, const float* __restrict__ proj,
               const float* __restrict__ g, const float* __restrict__ bta,
               float* __restrict__ out)
{
  __shared__ float r1[128];
  __shared__ float r2[128];
  const int row = blockIdx.x, tid = threadIdx.x;
  float x[4], s = 0.f, sq = 0.f;
#pragma unroll
  for (int i = 0; i < 4; ++i) {
    int c = tid + 128 * i;
    float v = enc[(size_t)row * 512 + c] + proj[(size_t)row * 512 + c];
    x[i] = v; s += v; sq += v * v;
  }
  r1[tid] = s; r2[tid] = sq; __syncthreads();
  for (int st = 64; st > 0; st >>= 1) {
    if (tid < st) { r1[tid] += r1[tid + st]; r2[tid] += r2[tid + st]; }
    __syncthreads();
  }
  const float mean = r1[0] * (1.0f / 512.0f);
  const float var  = r2[0] * (1.0f / 512.0f) - mean * mean;
  const float inv  = rsqrtf(var + 1e-5f);
#pragma unroll
  for (int i = 0; i < 4; ++i) {
    int c = tid + 128 * i;
    out[(size_t)row * 512 + c] = (x[i] - mean) * inv * g[c] + bta[c];
  }
}

// ---------------------------------------------------------------------------
extern "C" void kernel_launch(void* const* d_in, const int* in_sizes, int n_in,
                              void* d_out, int out_size, void* d_ws, size_t ws_size,
                              hipStream_t stream)
{
  (void)in_sizes; (void)n_in; (void)out_size; (void)ws_size;
  const float* ctx   = (const float*)d_in[0];   // (2000,512)
  const float* enc   = (const float*)d_in[1];   // (8,1024,512)
  const float* wq    = (const float*)d_in[2];
  const float* bq    = (const float*)d_in[3];
  const float* wk    = (const float*)d_in[4];
  const float* bk    = (const float*)d_in[5];
  const float* wv    = (const float*)d_in[6];
  const float* bv    = (const float*)d_in[7];
  const float* wo    = (const float*)d_in[8];
  const float* bo    = (const float*)d_in[9];
  const float* wcomb = (const float*)d_in[10];
  const float* bcomb = (const float*)d_in[11];
  const float* lng   = (const float*)d_in[12];
  const float* lnb   = (const float*)d_in[13];
  float* out = (float*)d_out;

  // workspace layout (~67 MB)
  float* ws  = (float*)d_ws;
  float* q   = ws;                                 // 8192*512
  float* kc  = q   + (size_t)8192 * 512;           // 2000*512
  float* vc  = kc  + (size_t)2000 * 512;           // 2000*512
  float* S1  = vc  + (size_t)2000 * 512;           // 1024*2000
  float* rmx = S1  + (size_t)1024 * 2000;          // 1024
  float* rsm = rmx + 1024;                         // 1024
  float* csc = rsm + 1024;                         // 2048 (2000 used)
  float* o   = csc + 2048;                         // 8192*512
  float* o3  = o   + (size_t)8192 * 512;           // 8192*512
  int*   idxb = (int*)(o3 + (size_t)8192 * 512);   // 128
  unsigned short* k2f = (unsigned short*)(idxb + 128); // 112*512 bf16
  unsigned short* v2f = k2f + (size_t)112 * 512;       // 112*512 bf16
  float* o2 = q;   // q is dead after attn2 -> reuse for second projection

  const float scale = 0.0883883476483184f;
  const dim3 blk(128);

  init_zero<<<dim3(8), dim3(256), 0, stream>>>(csc, 2048);

  // Q/K/V projections (bf16 WMMA GEMMs, 64x64 block tiles)
  gemm_nt_wmma<<<dim3(8, 128), blk, 0, stream>>>(enc, 512, wq, 512, bq, q,  512, 8192, 512, 512, 1.0f);
  gemm_nt_wmma<<<dim3(8, 32),  blk, 0, stream>>>(ctx, 512, wk, 512, bk, kc, 512, 2000, 512, 512, 1.0f);
  gemm_nt_wmma<<<dim3(8, 32),  blk, 0, stream>>>(ctx, 512, wv, 512, bv, vc, 512, 2000, 512, 512, 1.0f);

  // stage-1 scores for b=0 only; accumulate softmax column-sums per head
  for (int h = 0; h < 4; ++h) {
    gemm_nt_wmma<<<dim3(32, 16), blk, 0, stream>>>(q + h * 128, 512, kc + h * 128, 512,
                                                   nullptr, S1, 2000, 1024, 2000, 128, scale);
    rowstats_kernel<<<dim3(1024), dim3(256), 0, stream>>>(S1, 2000, rmx, rsm);
    colsum_kernel<<<dim3(8), dim3(256), 0, stream>>>(S1, 1024, 2000, rmx, rsm, csc);
  }

  topk_kernel<<<dim3(1), dim3(256), 0, stream>>>(csc, 2000, 101, idxb);
  gather_kernel<<<dim3(28), dim3(256), 0, stream>>>(kc, vc, idxb, k2f, v2f);

  // fused stage-2 attention (flash-style, WMMA for S and P·V, async LDS staging)
  attn2_kernel<<<dim3(64, 4, 8), dim3(32), 0, stream>>>(q, k2f, v2f, o);

  // output projections + residual LayerNorm
  gemm_nt_wmma<<<dim3(8, 128), blk, 0, stream>>>(o,  512, wo,    512, bo,    o2, 512, 8192, 512, 512, 1.0f);
  gemm_nt_wmma<<<dim3(8, 128), blk, 0, stream>>>(o2, 512, wcomb, 512, bcomb, o3, 512, 8192, 512, 512, 1.0f);
  ln_kernel<<<dim3(8192), dim3(128), 0, stream>>>(enc, o3, lng, lnb, out);
}
